// DeepAR_36498632081747
// MI455X (gfx1250) — compile-verified, matrix operand order checked by reading, not run on hardware
//
#include <hip/hip_runtime.h>

typedef __attribute__((ext_vector_type(16))) _Float16 v16h;
typedef __attribute__((ext_vector_type(8)))  float    v8f;

#define TBATCH 32
#define TN     512
#define TL_IN  336
#define TL_OUT 48
#define TSTEPS 383          // L_IN + L_OUT - 1

#define W0_STRIDE 96        // [prev,cov4,bias1,pad...32] | W_hh0 (64)
#define W1_STRIDE 128       // W_ih1 (64) | W_hh1 (64)
#define H_STRIDE  72        // 64 cols + pad (bank spread)

// LDS: halves(256*96 + 256*128 + 2*128*72 + 16*64)=76800 -> 153600B
// + 256 f32 bias1 + 2 f32 head_b
#define SMEM_BYTES (153600 + 256*4 + 16)

// Branch-free activations built from TRANS ops only (v_exp_f32 + v_rcp_f32),
// no exec divergence and no IEEE div expansion (v_div_scale/v_div_fmas).
__device__ __forceinline__ float sigm(float x) {
  return __builtin_amdgcn_rcpf(1.f + __expf(-x));
}
__device__ __forceinline__ float fast_tanh(float x) {
  // tanh(x) = 2*sigmoid(2x) - 1 ; saturates correctly via inf -> rcp -> 0
  return 2.f * __builtin_amdgcn_rcpf(1.f + __expf(-2.f * x)) - 1.f;
}

// Load one 16x32 f16 A/B fragment chunk from a row-major [row][K] LDS row.
// lane<16 holds K {0..7, 16..23}; lane>=16 holds K {8..15, 24..31}.
__device__ __forceinline__ v16h ldfrag(const _Float16* p, int hi) {
  union { v16h v; float4 f[2]; } u;
  const float4* q = reinterpret_cast<const float4*>(p);
  u.f[0] = q[hi];
  u.f[1] = q[2 + hi];
  return u.v;
}

#define WMMA(a, b, c) \
  __builtin_amdgcn_wmma_f32_16x16x32_f16(false, (a), false, (b), (short)0, (c), false, false)

__global__ void __launch_bounds__(256, 1)
deepar_lstm_wmma(const float* __restrict__ hist, const float* __restrict__ fut,
                 const float* __restrict__ embW, const float* __restrict__ embB,
                 const float* __restrict__ Wih0, const float* __restrict__ Whh0,
                 const float* __restrict__ bih0, const float* __restrict__ bhh0,
                 const float* __restrict__ Wih1, const float* __restrict__ Whh1,
                 const float* __restrict__ bih1, const float* __restrict__ bhh1,
                 const float* __restrict__ headW, const float* __restrict__ headB,
                 float* __restrict__ out) {
  extern __shared__ float4 smem_raw4[];
  _Float16* sW0 = reinterpret_cast<_Float16*>(smem_raw4);
  _Float16* sW1 = sW0 + 256 * W0_STRIDE;
  _Float16* sH0 = sW1 + 256 * W1_STRIDE;
  _Float16* sH1 = sH0 + 128 * H_STRIDE;
  _Float16* sHW = sH1 + 128 * H_STRIDE;       // 16 x 64, rows 2..15 zero
  float*    sB1 = reinterpret_cast<float*>(sHW + 16 * 64);
  float*    sHB = sB1 + 256;

  // ---- one-time weight fold + f16 conversion into LDS -------------------
  for (int n = threadIdx.x; n < 256; n += blockDim.x) {
    const float* wr = Wih0 + n * 36;           // 32 emb cols + 4 cov cols
    float dp = 0.f, db = 0.f;
    for (int k = 0; k < 32; ++k) { dp += wr[k] * embW[k]; db += wr[k] * embB[k]; }
    _Float16* d = sW0 + n * W0_STRIDE;
    d[0] = (_Float16)dp;                       // coeff of prev (rank-1 fold)
    d[1] = (_Float16)wr[32]; d[2] = (_Float16)wr[33];
    d[3] = (_Float16)wr[34]; d[4] = (_Float16)wr[35];
    d[5] = (_Float16)(bih0[n] + bhh0[n] + db); // bias via constant-1 column
    for (int k = 6; k < 32; ++k) d[k] = (_Float16)0.f;
    const float* hr = Whh0 + n * 64;
    for (int k = 0; k < 64; ++k) d[32 + k] = (_Float16)hr[k];

    _Float16* d1 = sW1 + n * W1_STRIDE;
    const float* i1 = Wih1 + n * 64;
    const float* h1w = Whh1 + n * 64;
    for (int k = 0; k < 64; ++k) { d1[k] = (_Float16)i1[k]; d1[64 + k] = (_Float16)h1w[k]; }
    sB1[n] = bih1[n] + bhh1[n];
  }
  for (int i = threadIdx.x; i < 16 * 64; i += blockDim.x) {
    int r = i >> 6, k = i & 63;
    sHW[i] = (r < 2) ? (_Float16)headW[r * 64 + k] : (_Float16)0.f;
  }
  for (int i = threadIdx.x; i < 128 * H_STRIDE; i += blockDim.x) {
    sH0[i] = (_Float16)0.f; sH1[i] = (_Float16)0.f;
  }
  if (threadIdx.x < 2) sHB[threadIdx.x] = headB[threadIdx.x];
  __syncthreads();

  // ---- per-wave persistent state ---------------------------------------
  const int lane = threadIdx.x & 31;
  const int wave = threadIdx.x >> 5;
  const int hi   = lane >> 4;
  const int col  = lane & 15;
  const int wrow = wave * 16;
  const int seqBase = blockIdx.x * 128 + wrow;

  float c0s[4][8] = {};
  float c1s[4][8] = {};
  float bias1v[16];
#pragma unroll
  for (int nt = 0; nt < 16; ++nt) bias1v[nt] = sB1[nt * 16 + col];

  const v16h bh0 = ldfrag(sHW + col * 64 + 0, hi);
  const v16h bh1 = ldfrag(sHW + col * 64 + 32, hi);

  const _Float16* arow0 = sH0 + (wrow + col) * H_STRIDE;
  const _Float16* arow1 = sH1 + (wrow + col) * H_STRIDE;

#pragma unroll 1
  for (int t = 0; t < TSTEPS; ++t) {
    // -------- input fragment: [prev, cov0..3, 1, 0...] (K=32, 6 live) ----
    union { v16h v; _Float16 h[16]; } ax;
#pragma unroll
    for (int i = 0; i < 16; ++i) ax.h[i] = (_Float16)0.f;
    if (lane < 16) {
      int seq = seqBase + lane;
      int b = seq >> 9, n = seq & 511;
      float prev;
      if (t < TL_IN) prev = hist[(((size_t)b * TL_IN + t) * TN + n) * 5];
      else           prev = fut[(((size_t)b * TL_OUT + (t - TL_IN)) * TN + n) * 5];
      int tc = t + 1;
      const float* cp = (tc < TL_IN)
          ? (hist + (((size_t)b * TL_IN + tc) * TN + n) * 5 + 1)
          : (fut  + (((size_t)b * TL_OUT + (tc - TL_IN)) * TN + n) * 5 + 1);
      ax.h[0] = (_Float16)prev;
      ax.h[1] = (_Float16)cp[0]; ax.h[2] = (_Float16)cp[1];
      ax.h[3] = (_Float16)cp[2]; ax.h[4] = (_Float16)cp[3];
      ax.h[5] = (_Float16)1.f;
    }

    // previous-step hidden states as A fragments
    v16h ah0a = ldfrag(arow0 + 0, hi);
    v16h ah0b = ldfrag(arow0 + 32, hi);
    v16h ah1a = ldfrag(arow1 + 0, hi);
    v16h ah1b = ldfrag(arow1 + 32, hi);

    // -------- layer 0: two h-tile groups at a time (8 live accumulators,
    // 64 VGPRs) -> 8 independent WMMA chains so the scheduler keeps
    // B-fragment ds_loads in flight ahead of the consuming WMMAs.
#pragma unroll
    for (int jp = 0; jp < 4; jp += 2) {
      v8f acc[2][4];
#pragma unroll
      for (int jj = 0; jj < 2; ++jj) {
#pragma unroll
        for (int q = 0; q < 4; ++q) {
          const _Float16* wr = sW0 + ((q * 4 + jp + jj) * 16 + col) * W0_STRIDE;
          v8f a = {0.f, 0.f, 0.f, 0.f, 0.f, 0.f, 0.f, 0.f};
          a = WMMA(ax.v, ldfrag(wr + 0, hi), a);
          a = WMMA(ah0a, ldfrag(wr + 32, hi), a);
          a = WMMA(ah0b, ldfrag(wr + 64, hi), a);
          acc[jj][q] = a;
        }
      }
#pragma unroll
      for (int jj = 0; jj < 2; ++jj) {
        int j = jp + jj;
#pragma unroll
        for (int r = 0; r < 8; ++r) {
          float iv = sigm(acc[jj][0][r]);
          float fv = sigm(acc[jj][1][r]);
          float gv = fast_tanh(acc[jj][2][r]);
          float ov = sigm(acc[jj][3][r]);
          float c = fv * c0s[j][r] + iv * gv;
          c0s[j][r] = c;
          float h = ov * fast_tanh(c);
          int m = hi ? (r + 8) : r;
          sH0[(wrow + m) * H_STRIDE + j * 16 + col] = (_Float16)h;  // D->A transpose via LDS
        }
      }
    }
    v16h ah0na = ldfrag(arow0 + 0, hi);    // fresh h0 of this step
    v16h ah0nb = ldfrag(arow0 + 32, hi);

    // -------- layer 1: two h-tile groups at a time, K=128 [h0 | h1] ------
#pragma unroll
    for (int jp = 0; jp < 4; jp += 2) {
      v8f acc[2][4];
#pragma unroll
      for (int jj = 0; jj < 2; ++jj) {
#pragma unroll
        for (int q = 0; q < 4; ++q) {
          int nt = q * 4 + jp + jj;
          const _Float16* wr = sW1 + (nt * 16 + col) * W1_STRIDE;
          float bv = bias1v[nt];
          v8f a = {bv, bv, bv, bv, bv, bv, bv, bv};
          a = WMMA(ah0na, ldfrag(wr + 0, hi), a);
          a = WMMA(ah0nb, ldfrag(wr + 32, hi), a);
          a = WMMA(ah1a, ldfrag(wr + 64, hi), a);
          a = WMMA(ah1b, ldfrag(wr + 96, hi), a);
          acc[jj][q] = a;
        }
      }
#pragma unroll
      for (int jj = 0; jj < 2; ++jj) {
        int j = jp + jj;
#pragma unroll
        for (int r = 0; r < 8; ++r) {
          float iv = sigm(acc[jj][0][r]);
          float fv = sigm(acc[jj][1][r]);
          float gv = fast_tanh(acc[jj][2][r]);
          float ov = sigm(acc[jj][3][r]);
          float c = fv * c1s[j][r] + iv * gv;
          c1s[j][r] = c;
          float h = ov * fast_tanh(c);
          int m = hi ? (r + 8) : r;
          sH1[(wrow + m) * H_STRIDE + j * 16 + col] = (_Float16)h;
        }
      }
    }

    // -------- head: relu(h1) @ head_W^T via one padded N-tile ------------
    union { v16h v; _Float16 h[16]; } ar0u, ar1u;
    ar0u.v = ldfrag(arow1 + 0, hi);
    ar1u.v = ldfrag(arow1 + 32, hi);
#pragma unroll
    for (int i = 0; i < 16; ++i) {
      ar0u.h[i] = (ar0u.h[i] > (_Float16)0.f) ? ar0u.h[i] : (_Float16)0.f;
      ar1u.h[i] = (ar1u.h[i] > (_Float16)0.f) ? ar1u.h[i] : (_Float16)0.f;
    }
    v8f p = {0.f, 0.f, 0.f, 0.f, 0.f, 0.f, 0.f, 0.f};
    p = WMMA(ar0u.v, bh0, p);
    p = WMMA(ar1u.v, bh1, p);

    if (t >= TSTEPS - TL_OUT) {
      int ch = -1;
      if (lane == 0 || lane == 16) ch = 0;          // mu column
      else if (lane == 1 || lane == 17) ch = 1;     // sigma column
      if (ch >= 0) {
        float bv = sHB[ch];
        int tp = t - (TSTEPS - TL_OUT);
#pragma unroll
        for (int r = 0; r < 8; ++r) {
          int seq = seqBase + (hi ? (r + 8) : r);
          int b = seq >> 9, n = seq & 511;
          float v = p[r] + bv;
          if (ch == 1) {
            // branch-free stable softplus: max(v,0) + log1p(exp(-|v|))
            v = fmaxf(v, 0.f) + __logf(1.f + __expf(-fabsf(v)));
          }
          out[(((size_t)b * TL_OUT + tp) * TN + n) * 2 + ch] = v;
        }
      }
    }
  }
}

extern "C" void kernel_launch(void* const* d_in, const int* in_sizes, int n_in,
                              void* d_out, int out_size, void* d_ws, size_t ws_size,
                              hipStream_t stream) {
  const float* hist  = (const float*)d_in[0];
  const float* fut   = (const float*)d_in[1];
  const float* embW  = (const float*)d_in[2];
  const float* embB  = (const float*)d_in[3];
  const float* Wih0  = (const float*)d_in[4];
  const float* Whh0  = (const float*)d_in[5];
  const float* bih0  = (const float*)d_in[6];
  const float* bhh0  = (const float*)d_in[7];
  const float* Wih1  = (const float*)d_in[8];
  const float* Whh1  = (const float*)d_in[9];
  const float* bih1  = (const float*)d_in[10];
  const float* bhh1  = (const float*)d_in[11];
  const float* headW = (const float*)d_in[12];
  const float* headB = (const float*)d_in[13];

  (void)hipFuncSetAttribute(reinterpret_cast<const void*>(deepar_lstm_wmma),
                            hipFuncAttributeMaxDynamicSharedMemorySize, SMEM_BYTES);

  // 16384 sequences / (8 waves * 16 rows) = 128 blocks
  deepar_lstm_wmma<<<128, 256, SMEM_BYTES, stream>>>(
      hist, fut, embW, embB, Wih0, Whh0, bih0, bhh0,
      Wih1, Whh1, bih1, bhh1, headW, headB, (float*)d_out);
}
